// MultiHeadAttention_20538533610261
// MI455X (gfx1250) — matrix operand order repaired and checked by reading, not compiled, not run on hardware
//
#include <hip/hip_runtime.h>
#include <hip/hip_bf16.h>

// ---------------------------------------------------------------------------
// MHA for MI455X (gfx1250, wave32). All GEMMs on v_wmma_f32_16x16x32_bf16.
// Tile staging of row-major tiles via Tensor Data Mover (tensor_load_to_lds,
// TENSORcnt), transposed tiles staged manually. D=1024,H=16,HD=64,B=2,S=2048.
// ---------------------------------------------------------------------------

#define DMODEL 1024
#define NHEAD  16
#define HDIM   64
#define BSZ    2
#define SEQ    2048
#define MTOT   (BSZ * SEQ)   // 4096

typedef __attribute__((ext_vector_type(16))) __bf16 v16bf;
typedef __attribute__((ext_vector_type(8)))  float  v8f;
typedef __attribute__((ext_vector_type(4)))  unsigned int v4u;
typedef __attribute__((ext_vector_type(8)))  int v8i;
typedef __attribute__((ext_vector_type(4)))  int v4i;

union Frag {
  v16bf v;
  uint4 q[2];
  unsigned short h[16];
};

__device__ __forceinline__ unsigned short f2bf(float f) {
  unsigned int u = __float_as_uint(f);
  u += 0x7FFFu + ((u >> 16) & 1u);   // round-to-nearest-even
  return (unsigned short)(u >> 16);
}
__device__ __forceinline__ float bf2f(unsigned short h) {
  return __uint_as_float(((unsigned int)h) << 16);
}
__device__ __forceinline__ v8f vzero8() {
  v8f z;
  for (int i = 0; i < 8; ++i) z[i] = 0.f;
  return z;
}

// --------------------------- TDM 2-D tile load -----------------------------
// D# per CDNA5 ISA. bf16 (data_size=2B) 2-D tile of
// tile_d0 x tile_d1 elements from row-major global (row stride stride_e
// elements) into LDS at lds_off, padding pad_amt_code (+1 DWORDs) after every
// 2^(pad_int_code+1) DWORDs stored. Groups 2/3 zero (<=2D tensor).
// 6-arg builtin form (clang-23 / therock-10.0 toolchain).
__device__ __forceinline__ void tdm_load_2d(unsigned lds_off, const void* gaddr,
                                            unsigned tile_d0, unsigned tile_d1,
                                            unsigned stride_e,
                                            unsigned tensor_d0,
                                            unsigned pad_int_code,
                                            unsigned pad_amt_code) {
  unsigned long long ga = (unsigned long long)(uintptr_t)gaddr;
  const unsigned tensor_d1 = 1u << 20;
  v4u g0;
  g0[0] = 1u;                                    // count=1, user mode
  g0[1] = lds_off;                               // lds_addr
  g0[2] = (unsigned)(ga & 0xFFFFFFFFu);          // global_addr[31:0]
  g0[3] = (unsigned)((ga >> 32) & 0x1FFFFFFu)    // global_addr[56:32]
          | (2u << 30);                          // type=2 ("image")
  v8i g1;
  g1[0] = (int)((1u << 16)                       // data_size: 2 bytes
                | (1u << 20)                     // pad_enable
                | (pad_int_code << 22)           // pad_interval
                | (pad_amt_code << 25));         // pad_amount
  g1[1] = (int)((tensor_d0 & 0xFFFFu) << 16);    // dim0[15:0] @ bits63:48
  g1[2] = (int)((tensor_d0 >> 16) | ((tensor_d1 & 0xFFFFu) << 16));
  g1[3] = (int)((tensor_d1 >> 16) | (tile_d0 << 16));   // tile_dim0 @127:112
  g1[4] = (int)(tile_d1 & 0xFFFFu);              // tile_dim1; tile_dim2=0
  g1[5] = (int)stride_e;                         // dim0_stride[31:0]
  g1[6] = 0;                                     // dim0_stride[47:32]=0
  g1[7] = 0;                                     // dim1_stride=0 (2D)
  v4i gz4;
  gz4[0] = 0; gz4[1] = 0; gz4[2] = 0; gz4[3] = 0;
  v8i gz8;
  for (int i = 0; i < 8; ++i) gz8[i] = 0;
  __builtin_amdgcn_tensor_load_to_lds(g0, g1, gz4, gz4, gz8, 0);
}

// -------------------------- conversion kernels -----------------------------

__global__ void convert_f32_to_bf16(const float* __restrict__ in,
                                    unsigned short* __restrict__ out, int n) {
  int i = blockIdx.x * blockDim.x + threadIdx.x;
  if (i < n) out[i] = f2bf(in[i]);
}

// wT[d][e] = w[e][d]   (D x D); reference does out[s,e] = sum_d x[s,d]*w[e,d]
__global__ void transpose_to_bf16(const float* __restrict__ w,
                                  unsigned short* __restrict__ wT) {
  int i = blockIdx.x * blockDim.x + threadIdx.x;  // over D*D outputs
  int d = i >> 10, e = i & (DMODEL - 1);
  wT[(d << 10) + e] = f2bf(w[(e << 10) + d]);
}

// ------------------------------- GEMM --------------------------------------
// C[M,N] (f32 acc) = A[M,K](bf16 row-major) * Bt[K,N](bf16 row-major, already
// transposed weights).  Block tile 128x128, 8 waves, K-steps of 32.
// A tile staged by TDM (row pitch 64B+16B pad = 80B); B tile staged manually
// (needs transpose).  MODE 0: scatter bf16 into [B,H,S,HD] head layout.
// MODE 1: f32 row-major.

template <int MODE>
__global__ __launch_bounds__(256) void gemm_bf16(
    const unsigned short* __restrict__ A, const unsigned short* __restrict__ Bt,
    void* __restrict__ Cout) {
  constexpr int K = DMODEL, N = DMODEL;
  constexpr int AP = 40;  // LDS pitch (elems): 80B rows, b128-aligned, conflict-free
  constexpr int BP = 40;
  __shared__ unsigned short As[128 * AP];   // As[m][k]
  __shared__ unsigned short Bs[128 * BP];   // Bs[n][k] (transposed tile)

  const int tid = threadIdx.x;
  const int wave = tid >> 5, lane = tid & 31;
  const int l15 = lane & 15, hh = lane >> 4;
  const int m0 = blockIdx.y * 128, n0 = blockIdx.x * 128;
  const int mywave = __builtin_amdgcn_readfirstlane(tid) >> 5;  // scalar branch
  const unsigned as_off = (unsigned)(uintptr_t)(&As[0]);

  v8f acc[8];
  for (int i = 0; i < 8; ++i) acc[i] = vzero8();

  for (int k0 = 0; k0 < K; k0 += 32) {
    __syncthreads();
    // ---- A tile via Tensor Data Mover: 128x32 bf16, stride K, padded rows
    if (mywave == 0) {
      tdm_load_2d(as_off, A + (size_t)m0 * K + k0,
                  /*tile_d0=*/32, /*tile_d1=*/128, /*stride_e=*/K,
                  /*tensor_d0=*/K, /*pad_int=*/3 /*16 DW*/,
                  /*pad_amt=*/3 /*4 DW*/);
    }
    // ---- B tile manual (transpose): Bs[n][k] = Bt[k0+k][n0+n]
    {
      int kk = tid >> 3, nn = (tid & 7) * 16;
      const uint4* bsrc = (const uint4*)(Bt + (size_t)(k0 + kk) * N + n0 + nn);
      uint4 b0 = bsrc[0], b1 = bsrc[1];
      if (k0 + 32 < K)  // cover next tile's latency
        __builtin_prefetch(Bt + (size_t)(k0 + 32 + kk) * N + n0 + nn, 0, 1);
      const unsigned short* bh = (const unsigned short*)&b0;
      for (int j = 0; j < 8; ++j) Bs[(nn + j) * BP + kk] = bh[j];
      bh = (const unsigned short*)&b1;
      for (int j = 0; j < 8; ++j) Bs[(nn + 8 + j) * BP + kk] = bh[j];
    }
    if (mywave == 0) __builtin_amdgcn_s_wait_tensorcnt(0);
    __syncthreads();

    // A fragment, 16x32: lane(l15,hh) holds A[m=l15][8h+i | 16+8h+i]
    Frag af;
    const unsigned short* arow = As + (wave * 16 + l15) * AP;
    af.q[0] = *(const uint4*)(arow + 8 * hh);
    af.q[1] = *(const uint4*)(arow + 16 + 8 * hh);

    for (int nt = 0; nt < 8; ++nt) {
      // B fragment, 32x16: lane holds B[k=8h+i | 16+8h+i][n=l15]
      Frag bf;
      const unsigned short* brow = Bs + (nt * 16 + l15) * BP;
      bf.q[0] = *(const uint4*)(brow + 8 * hh);
      bf.q[1] = *(const uint4*)(brow + 16 + 8 * hh);
      acc[nt] = __builtin_amdgcn_wmma_f32_16x16x32_bf16(
          false, af.v, false, bf.v, (short)0, acc[nt], false, false);
    }
  }

  // epilogue: C[m = r+8h][n = l15] in acc[nt][r]
  if (MODE == 0) {
    unsigned short* dst = (unsigned short*)Cout;
    for (int nt = 0; nt < 8; ++nt) {
      int gcol = n0 + nt * 16 + l15;
      int head = gcol >> 6, hd = gcol & 63;
      for (int r = 0; r < 8; ++r) {
        int gm = m0 + wave * 16 + r + 8 * hh;
        int b = gm >> 11, s = gm & (SEQ - 1);
        dst[(((size_t)(b * NHEAD + head) * SEQ + s) << 6) + hd] =
            f2bf(acc[nt][r]);
      }
    }
  } else {
    float* dst = (float*)Cout;
    for (int nt = 0; nt < 8; ++nt) {
      int gcol = n0 + nt * 16 + l15;
      for (int r = 0; r < 8; ++r) {
        int gm = m0 + wave * 16 + r + 8 * hh;
        dst[(size_t)gm * N + gcol] = acc[nt][r];
      }
    }
  }
}

// ------------------------------- RoPE --------------------------------------
// t: [B*H, S, 64] bf16, in-place. scale folds 1/sqrt(HD) into q.
__global__ void rope_kernel(unsigned short* __restrict__ t, float scale) {
  int tid = blockIdx.x * blockDim.x + threadIdx.x;  // B*H*S*32 threads
  int j = tid & 31;
  int row = tid >> 5;
  int s = row & (SEQ - 1);
  // inv_freq[j] = 10000^(-j/32) = exp(-ln(1e4)/32 * j)
  float ang = (float)s * __expf(-0.28782313662425572f * (float)j);
  float c, sn;
  __sincosf(ang, &sn, &c);
  unsigned short* p = t + ((size_t)row << 6);
  float x0 = bf2f(p[j]), x1 = bf2f(p[j + 32]);
  p[j]      = f2bf((x0 * c - x1 * sn) * scale);
  p[j + 32] = f2bf((x1 * c + x0 * sn) * scale);
}

// --------------------------- flash attention -------------------------------
// grid (S/128, B*H), 256 threads. Wave w owns q rows [qc+16w, qc+16w+16).
// K tile staged by TDM (row 128B + 16B pad = 144B pitch); V staged manually
// transposed.
__global__ __launch_bounds__(256) void attn_kernel(
    const unsigned short* __restrict__ qg, const unsigned short* __restrict__ kg,
    const unsigned short* __restrict__ vg, unsigned short* __restrict__ ctx) {
  constexpr int KP = 72;  // K tile pitch (elems): 144B rows, conflict-free
  constexpr int VP = 40;  // V^T tile pitch
  constexpr int PP = 40;  // per-wave P buffer pitch
  __shared__ unsigned short Ks[32 * KP];       // Ks[key][d]
  __shared__ unsigned short VTs[64 * VP];      // VTs[d][key]
  __shared__ unsigned short Pls[8 * 16 * PP];  // per-wave P 16x32

  const int tid = threadIdx.x, wave = tid >> 5, lane = tid & 31;
  const int l15 = lane & 15, hh = lane >> 4;
  const int bh = blockIdx.y;            // b*16 + h
  const int qc = blockIdx.x * 128;
  const size_t base = (size_t)bh * SEQ * HDIM;
  const int mywave = __builtin_amdgcn_readfirstlane(tid) >> 5;
  const unsigned ks_off = (unsigned)(uintptr_t)(&Ks[0]);

  // Q fragments (A operand) for K-halves [0,32) and [32,64)
  Frag qf0, qf1;
  {
    const unsigned short* qp = qg + base + (size_t)(qc + wave * 16 + l15) * HDIM;
    qf0.q[0] = *(const uint4*)(qp + 8 * hh);
    qf0.q[1] = *(const uint4*)(qp + 16 + 8 * hh);
    qf1.q[0] = *(const uint4*)(qp + 32 + 8 * hh);
    qf1.q[1] = *(const uint4*)(qp + 48 + 8 * hh);
  }

  float mrow[8], lrow[8];
  v8f acco[4];
  for (int r = 0; r < 8; ++r) { mrow[r] = -1e30f; lrow[r] = 0.f; }
  for (int nt = 0; nt < 4; ++nt) acco[nt] = vzero8();

  const int qlast = qc + wave * 16 + 15;
  const int nkb = (qc >> 5) + 4;   // 32-key blocks covering [0, qc+128)

  for (int kb = 0; kb < nkb; ++kb) {
    const int key0 = kb * 32;
    __syncthreads();
    // ---- K tile via TDM: 32 rows x 64 elems, padded to 144B pitch
    if (mywave == 0) {
      tdm_load_2d(ks_off, kg + base + (size_t)key0 * HDIM,
                  /*tile_d0=*/64, /*tile_d1=*/32, /*stride_e=*/64,
                  /*tensor_d0=*/64, /*pad_int=*/4 /*32 DW*/,
                  /*pad_amt=*/3 /*4 DW*/);
    }
    // ---- V tile manual transpose: VTs[d][key]
    {
      int key = tid >> 3, seg = (tid & 7) * 8;
      uint4 vv = *(const uint4*)(vg + base + (size_t)(key0 + key) * HDIM + seg);
      const unsigned short* vh = (const unsigned short*)&vv;
      for (int j = 0; j < 8; ++j) VTs[(seg + j) * VP + key] = vh[j];
    }
    if (mywave == 0) __builtin_amdgcn_s_wait_tensorcnt(0);
    __syncthreads();
    if (key0 > qlast) continue;   // wave-uniform; all waves hit next barrier

    // ---- scores: two 16x16 tiles, each = Q(16x64) . K^T, 2 chained WMMAs
    float s0[8], s1[8];
    for (int kt = 0; kt < 2; ++kt) {
      Frag kf0, kf1;
      const unsigned short* krow = Ks + (kt * 16 + l15) * KP;
      kf0.q[0] = *(const uint4*)(krow + 8 * hh);
      kf0.q[1] = *(const uint4*)(krow + 16 + 8 * hh);
      kf1.q[0] = *(const uint4*)(krow + 32 + 8 * hh);
      kf1.q[1] = *(const uint4*)(krow + 48 + 8 * hh);
      v8f sacc = vzero8();
      sacc = __builtin_amdgcn_wmma_f32_16x16x32_bf16(
          false, qf0.v, false, kf0.v, (short)0, sacc, false, false);
      sacc = __builtin_amdgcn_wmma_f32_16x16x32_bf16(
          false, qf1.v, false, kf1.v, (short)0, sacc, false, false);
      int keyg = key0 + kt * 16 + l15;
      float* sd = kt ? s1 : s0;
      for (int r = 0; r < 8; ++r) {
        int qrow = qc + wave * 16 + r + 8 * hh;
        sd[r] = (keyg <= qrow) ? sacc[r] : -1e30f;  // causal mask
      }
    }

    // ---- online softmax (rows live on fixed 16-lane halves -> shfl_xor<=8)
    float pr0[8], pr1[8], cr[8];
    for (int r = 0; r < 8; ++r) {
      float mt = fmaxf(s0[r], s1[r]);
      for (int off = 8; off >= 1; off >>= 1)
        mt = fmaxf(mt, __shfl_xor(mt, off, 32));
      float mn = fmaxf(mrow[r], mt);
      float c = __expf(mrow[r] - mn);
      float p0 = __expf(s0[r] - mn);
      float p1 = __expf(s1[r] - mn);
      float rs = p0 + p1;
      for (int off = 8; off >= 1; off >>= 1) rs += __shfl_xor(rs, off, 32);
      lrow[r] = lrow[r] * c + rs;
      mrow[r] = mn;
      pr0[r] = p0; pr1[r] = p1; cr[r] = c;
    }
    for (int nt = 0; nt < 4; ++nt)
      for (int r = 0; r < 8; ++r) acco[nt][r] *= cr[r];

    // ---- P: C-layout -> A-layout via per-wave LDS (DS ops wave-ordered)
    unsigned short* pbase = Pls + wave * 16 * PP;
    for (int r = 0; r < 8; ++r) {
      pbase[(r + 8 * hh) * PP + l15]      = f2bf(pr0[r]);
      pbase[(r + 8 * hh) * PP + 16 + l15] = f2bf(pr1[r]);
    }
    __builtin_amdgcn_wave_barrier();
    Frag pf;
    pf.q[0] = *(const uint4*)(pbase + l15 * PP + 8 * hh);
    pf.q[1] = *(const uint4*)(pbase + l15 * PP + 16 + 8 * hh);

    // ---- P(16x32) . V(32x64): 4 WMMAs into f32 accumulators
    for (int nt = 0; nt < 4; ++nt) {
      Frag vf;
      const unsigned short* vrow = VTs + (nt * 16 + l15) * VP;
      vf.q[0] = *(const uint4*)(vrow + 8 * hh);
      vf.q[1] = *(const uint4*)(vrow + 16 + 8 * hh);
      acco[nt] = __builtin_amdgcn_wmma_f32_16x16x32_bf16(
          false, pf.v, false, vf.v, (short)0, acco[nt], false, false);
    }
  }

  // epilogue: ctx[b, s, h*64 + hd] bf16
  const int b = bh >> 4, h = bh & 15;
  for (int nt = 0; nt < 4; ++nt) {
    int hd = nt * 16 + l15;
    for (int r = 0; r < 8; ++r) {
      int s = qc + wave * 16 + r + 8 * hh;
      float val = acco[nt][r] / lrow[r];
      ctx[((size_t)(b * SEQ + s) << 10) + (h << 6) + hd] = f2bf(val);
    }
  }
}

// ------------------------------- launch ------------------------------------

extern "C" void kernel_launch(void* const* d_in, const int* in_sizes, int n_in,
                              void* d_out, int out_size, void* d_ws,
                              size_t ws_size, hipStream_t stream) {
  const float* x  = (const float*)d_in[0];
  const float* wq = (const float*)d_in[1];
  const float* wk = (const float*)d_in[2];
  const float* wv = (const float*)d_in[3];
  const float* wo = (const float*)d_in[4];

  // workspace layout (bf16 elements), ~48 MB total
  unsigned short* ws  = (unsigned short*)d_ws;
  unsigned short* xb  = ws;              // 4,194,304  x in bf16 [B*S, D]
  unsigned short* wqT = xb  + 4194304;   // 1,048,576 each, [D, D] transposed
  unsigned short* wkT = wqT + 1048576;
  unsigned short* wvT = wkT + 1048576;
  unsigned short* woT = wvT + 1048576;
  unsigned short* qh  = woT + 1048576;   // 4,194,304 each, [B,H,S,64]
  unsigned short* kh  = qh  + 4194304;
  unsigned short* vh  = kh  + 4194304;
  unsigned short* ctx = vh  + 4194304;   // [B*S, D]

  convert_f32_to_bf16<<<4194304 / 256, 256, 0, stream>>>(x, xb, 4194304);
  transpose_to_bf16<<<1048576 / 256, 256, 0, stream>>>(wq, wqT);
  transpose_to_bf16<<<1048576 / 256, 256, 0, stream>>>(wk, wkT);
  transpose_to_bf16<<<1048576 / 256, 256, 0, stream>>>(wv, wvT);
  transpose_to_bf16<<<1048576 / 256, 256, 0, stream>>>(wo, woT);

  dim3 gg(DMODEL / 128, MTOT / 128);  // (8, 32)
  gemm_bf16<0><<<gg, 256, 0, stream>>>(xb, wqT, qh);
  gemm_bf16<0><<<gg, 256, 0, stream>>>(xb, wkT, kh);
  gemm_bf16<0><<<gg, 256, 0, stream>>>(xb, wvT, vh);

  rope_kernel<<<(BSZ * NHEAD * SEQ * 32) / 256, 256, 0, stream>>>(qh, 0.125f);
  rope_kernel<<<(BSZ * NHEAD * SEQ * 32) / 256, 256, 0, stream>>>(kh, 1.0f);

  attn_kernel<<<dim3(SEQ / 128, BSZ * NHEAD), 256, 0, stream>>>(qh, kh, vh, ctx);

  gemm_bf16<1><<<gg, 256, 0, stream>>>(ctx, woT, d_out);
}
